// GCRINT_52879637348792
// MI455X (gfx1250) — matrix operand order repaired and checked by reading, not compiled, not run on hardware
//
#include <hip/hip_runtime.h>
#include <hip/hip_bf16.h>

#define BB 8
#define TT 32
#define NN 144
#define HH 16
#define RCc 16
#define HNc 2304
#define G4c 9216
#define KT 72              // 2304 / 32 k-tiles
#define JT 576             // 9216 / 16 j-tiles

typedef __attribute__((ext_vector_type(16))) __bf16 v16bf;
typedef __attribute__((ext_vector_type(8)))  __bf16 v8bf;
typedef __attribute__((ext_vector_type(8)))  float  v8f;

__device__ __forceinline__ __bf16 f2bf(float f) {
  unsigned u = __builtin_bit_cast(unsigned, f);
  u = (u + 0x7FFFu + ((u >> 16) & 1u)) >> 16;   // round-to-nearest-even
  return __builtin_bit_cast(__bf16, (unsigned short)u);
}
__device__ __forceinline__ float sigm(float x) { return 1.0f / (1.0f + __expf(-x)); }

// ---- WMMA fragment helpers (layouts per CDNA5 ISA 7.12.2) ----

// A-matrix 16x32 bf16 from row-major source, leading dim ld.
// lanes 0-15: M=lane, K in {0..7,16..23}; lanes 16-31: M=lane-16, K in {8..15,24..31}.
__device__ __forceinline__ v16bf frag_a(const __bf16* base, int ld, int lane, int mvalid) {
  int m = lane & 15;
  if (m >= mvalid) {
    v16bf z;
#pragma unroll
    for (int e = 0; e < 16; ++e) z[e] = __builtin_bit_cast(__bf16, (unsigned short)0);
    return z;
  }
  const __bf16* p = base + (size_t)m * ld + ((lane < 16) ? 0 : 8);
  v8bf lo = *(const v8bf*)p;          // 16B aligned
  v8bf hi = *(const v8bf*)(p + 16);   // 16B aligned
  return __builtin_shufflevector(lo, hi, 0, 1, 2, 3, 4, 5, 6, 7, 8, 9, 10, 11, 12, 13, 14, 15);
}

// B-fragment from pre-swizzled tiled bf16 weights: one contiguous 32B load per lane.
__device__ __forceinline__ v16bf frag_bt(const __bf16* Wt, int jt, int kt, int lane) {
  return *(const v16bf*)(Wt + ((size_t)(jt * KT + kt) * 32 + lane) * 16);
}

__device__ __forceinline__ v8f wmma_bf16(v16bf a, v16bf b, v8f c) {
  return __builtin_amdgcn_wmma_f32_16x16x32_bf16(false, a, false, b, (short)0, c, false, false);
}

// ---- one-shot weight convert+swizzle: fp32 [9216 x 2304] row-major ->
//      bf16 tiled [jt][kt][lane][16], matching frag_bt exactly ----
__global__ void k_wconv(const float* W, __bf16* Wt) {
  int idx = blockIdx.x * blockDim.x + threadIdx.x;   // (jt, kt, lane)
  if (idx >= JT * KT * 32) return;
  int lane = idx & 31;
  int kt = (idx >> 5) % KT;
  int jt = idx / (32 * KT);
  const float* p = W + (size_t)(jt * 16 + (lane & 15)) * HNc + kt * 32 + ((lane < 16) ? 0 : 8);
  __bf16* q = Wt + (size_t)idx * 16;                 // 32B-aligned contiguous
#pragma unroll
  for (int e = 0; e < 16; ++e) q[e] = f2bf(p[e + ((e < 8) ? 0 : 8)]);
}

// ---- adjacency: adp = softmax(relu(nv1 @ nv2), axis=1) ----
__global__ void k_adp(const float* nv1, const float* nv2, float* adp) {
  int r = threadIdx.x;
  if (r >= NN) return;
  float mx = -1e30f;
  for (int w = 0; w < NN; ++w) {
    float s = 0.f;
    for (int k = 0; k < 10; ++k) s += nv1[r * 10 + k] * nv2[k * NN + w];
    s = s > 0.f ? s : 0.f;
    mx = fmaxf(mx, s);
  }
  float sum = 0.f;
  for (int w = 0; w < NN; ++w) {
    float s = 0.f;
    for (int k = 0; k < 10; ++k) s += nv1[r * 10 + k] * nv2[k * NN + w];
    s = s > 0.f ? s : 0.f;
    sum += __expf(s - mx);
  }
  float inv = 1.f / sum;
  for (int w = 0; w < NN; ++w) {
    float s = 0.f;
    for (int k = 0; k < 10; ++k) s += nv1[r * 10 + k] * nv2[k * NN + w];
    s = s > 0.f ? s : 0.f;
    adp[r * NN + w] = __expf(s - mx) * inv;
  }
}

// ---- features -> packed bf16 LSTM input, row = t*8+b, col = c*144+n ----
__global__ void k_featpack(const float* inp, const float* msk, const float* wsv, const float* bsv,
                           const float* wcv, const float* bcv, __bf16* Xin) {
  int idx = blockIdx.x * blockDim.x + threadIdx.x;
  if (idx >= BB * RCc * NN * TT) return;
  int t = idx & 31;
  int n = (idx >> 5) % NN;
  int c = (idx / (32 * NN)) & 15;
  int b = idx / (32 * NN * 16);
  float v = wsv[c] * inp[(b * TT + t) * NN + n] + bsv[c]
          + wcv[c] * msk[(b * TT + t) * NN + n] + bcv[c];
  Xin[(size_t)(t * BB + b) * HNc + c * NN + n] = f2bf(v);
}

// ---- Xp = Xin @ W.T + bih + bhh  (rows x 9216), WMMA bf16, tiled weights ----
// grid: (rows/16, 9216/64), block: 128 (4 waves, one 16x16 j-tile each)
// A-panel staged in LDS in two 36KB halves, shared by all 4 waves.
__global__ void __launch_bounds__(128) k_xproj(const __bf16* Xin, const __bf16* Wt,
                                               const float* bih, const float* bhh, float* Xp) {
  __shared__ __align__(16) __bf16 ash[16 * 1152];   // 36 KB half-panel
  int tid = threadIdx.x;
  int lane = tid & 31;
  int wave = tid >> 5;
  int mtile = blockIdx.x;
  int jt = blockIdx.y * 4 + wave;
  const __bf16* Ag = Xin + (size_t)(mtile * 16) * HNc;
  v8f acc;
#pragma unroll
  for (int r = 0; r < 8; ++r) acc[r] = 0.f;
  for (int half = 0; half < 2; ++half) {
    __syncthreads();
    const __bf16* src = Ag + half * 1152;
    for (int i = tid; i < 16 * 1152 / 8; i += 128) {
      int row = i / (1152 / 8);
      int col8 = i - row * (1152 / 8);
      ((uint4*)ash)[i] = *(const uint4*)(src + (size_t)row * HNc + col8 * 8);
    }
    __syncthreads();
#pragma unroll 4
    for (int kt = 0; kt < 36; ++kt) {
      v16bf a = frag_a(ash + kt * 32, 1152, lane, 16);
      v16bf b = frag_bt(Wt, jt, half * 36 + kt, lane);
      acc = wmma_bf16(a, b, acc);
    }
  }
  int n = lane & 15;
  int j = jt * 16 + n;
  float bias = bih[j] + bhh[j];
#pragma unroll
  for (int r = 0; r < 8; ++r) {
    int m = (lane < 16) ? r : r + 8;
    Xp[(size_t)(mtile * 16 + m) * G4c + j] = acc[r] + bias;
  }
}

// ---- one recurrent LSTM step, fused: h@Whh.T (WMMA) + gates + state update ----
// grid: 144 blocks (16 hidden units each), block: 128 (4 waves = 4 gates)
__global__ void __launch_bounds__(128) k_lstm_step(const __bf16* WhhT, const float* Xp, float* cst,
                                                   const __bf16* h_in, __bf16* h_out, float* hs,
                                                   int t, int Tt) {
  __shared__ __align__(16) __bf16 hsh[BB * HNc];  // 36 KB: h state (8x2304)
  __shared__ float gsh[4 * 8 * 16];               // 4 gates x 8 batch x 16 units
  int tid = threadIdx.x;
  {   // stage previous h into LDS (2304 x uint4)
    const uint4* s = (const uint4*)h_in;
    uint4* d = (uint4*)hsh;
    for (int i = tid; i < (BB * HNc) / 8; i += 128) d[i] = s[i];
  }
  __syncthreads();
  int lane = tid & 31;
  int g = tid >> 5;                 // this wave's gate (i,f,g,o)
  int jt = g * (HNc / 16) + blockIdx.x;
  v8f acc;
#pragma unroll
  for (int r = 0; r < 8; ++r) acc[r] = 0.f;
#pragma unroll 4
  for (int kt = 0; kt < KT; ++kt) {
    __builtin_prefetch((const void*)(WhhT + ((size_t)(jt * KT + kt + 2) * 32) * 16), 0, 1);
    v16bf a = frag_a(hsh + kt * 32, HNc, lane, 8);   // rows 8..15 padded zero
    v16bf b = frag_bt(WhhT, jt, kt, lane);
    acc = wmma_bf16(a, b, acc);
  }
  if (lane < 16) {                  // rows 0..7 = batches (upper half is padding)
#pragma unroll
    for (int r = 0; r < 8; ++r) gsh[(g * 8 + r) * 16 + lane] = acc[r];
  }
  __syncthreads();
  // 128 threads = 8 batches x 16 units: gate math + state update
  int b = tid >> 4, n = tid & 15;
  int u = blockIdx.x * 16 + n;
  const float* xp = Xp + (size_t)(t * BB + b) * G4c;
  float gi = gsh[(0 * 8 + b) * 16 + n] + xp[u];
  float gf = gsh[(1 * 8 + b) * 16 + n] + xp[HNc + u];
  float gg = gsh[(2 * 8 + b) * 16 + n] + xp[2 * HNc + u];
  float go = gsh[(3 * 8 + b) * 16 + n] + xp[3 * HNc + u];
  float cn = sigm(gf) * cst[b * HNc + u] + sigm(gi) * tanhf(gg);
  float hn = sigm(go) * tanhf(cn);
  cst[b * HNc + u] = cn;
  h_out[b * HNc + u] = f2bf(hn);
  hs[(size_t)(b * HNc + u) * Tt + t] = hn;
}

// ---- layer-0 combine fw/reversed-bw + tanh -> gcn input ----
__global__ void k_gcnin0(const float* hsA, const float* hsB, float* gin) {
  int idx = blockIdx.x * blockDim.x + threadIdx.x;
  if (idx >= BB * HNc * TT) return;
  int t = idx % TT;
  int rest = idx / TT;
  gin[idx] = tanhf(0.5f * (hsA[idx] + hsB[(size_t)rest * TT + (TT - 1 - t)]));
}

__global__ void k_gcnin1(const float* hsC, float* gin) {
  int idx = blockIdx.x * blockDim.x + threadIdx.x;
  if (idx >= BB * HNc * 16) return;
  gin[idx] = tanhf(hsC[idx]);
}

// ---- graph diffusion: y[b,c,w,t] = sum_v x[b,c,v,t] * A[v,w] ----
__global__ void k_diffuse(const float* x, const float* adp, float* y, int tl) {
  int idx = blockIdx.x * blockDim.x + threadIdx.x;
  int total = BB * HH * NN * tl;
  if (idx >= total) return;
  int t = idx % tl;
  int w = (idx / tl) % NN;
  int bc = idx / (tl * NN);
  const float* xb = x + (size_t)bc * NN * tl + t;
  float s = 0.f;
  for (int v = 0; v < NN; ++v) s += xb[(size_t)v * tl] * adp[v * NN + w];
  y[idx] = s;
}

// ---- 1x1 mix over [x;x1;x2] (3H=48 channels) -> RC channels ----
__global__ void k_gcmix(const float* x0, const float* x1, const float* x2,
                        const float* Wg, const float* bg, float* go, int tl) {
  int idx = blockIdx.x * blockDim.x + threadIdx.x;
  int total = BB * RCc * NN * tl;
  if (idx >= total) return;
  int t = idx % tl;
  int n = (idx / tl) % NN;
  int o = (idx / (tl * NN)) % RCc;
  int b = idx / (tl * NN * RCc);
  const float* wr = Wg + o * 48;
  size_t basei = (size_t)b * HH * NN * tl + (size_t)n * tl + t;
  float s = bg[o];
  for (int c = 0; c < HH; ++c) {
    size_t off = basei + (size_t)c * NN * tl;
    s += x0[off] * wr[c] + x1[off] * wr[16 + c] + x2[off] * wr[32 + c];
  }
  go[idx] = s;
}

// ---- batchnorm stats over go0[..., ::2] per channel ----
__global__ void k_bnstats(const float* go0, float* stats) {
  __shared__ float ssum[256], ssq[256];
  int c = blockIdx.x, tid = threadIdx.x;
  float s = 0.f, q = 0.f;
  for (int i = tid; i < BB * NN * 16; i += 256) {
    int t = i & 15;
    int n = (i >> 4) % NN;
    int b = i / (16 * NN);
    float v = go0[((size_t)(b * RCc + c) * NN + n) * TT + 2 * t];
    s += v; q += v * v;
  }
  ssum[tid] = s; ssq[tid] = q;
  __syncthreads();
  for (int st = 128; st > 0; st >>= 1) {
    if (tid < st) { ssum[tid] += ssum[tid + st]; ssq[tid] += ssq[tid + st]; }
    __syncthreads();
  }
  if (tid == 0) {
    float m = ssum[0] / (float)(BB * NN * 16);
    stats[c] = m;
    stats[RCc + c] = ssq[0] / (float)(BB * NN * 16) - m * m;
  }
}

// ---- bn-normalize + pack layer-1 LSTM input as bf16 ----
__global__ void k_bnpack(const float* go0, const float* stats, const float* gamma,
                         const float* beta, __bf16* XinC) {
  int idx = blockIdx.x * blockDim.x + threadIdx.x;
  if (idx >= BB * RCc * NN * 16) return;
  int t = idx & 15;
  int n = (idx >> 4) % NN;
  int c = (idx / (16 * NN)) & 15;
  int b = idx / (16 * NN * 16);
  float v = go0[((size_t)(b * RCc + c) * NN + n) * TT + 2 * t];
  float o = gamma[c] * (v - stats[c]) * rsqrtf(stats[RCc + c] + 1e-5f) + beta[c];
  XinC[(size_t)(t * BB + b) * HNc + c * NN + n] = f2bf(o);
}

// ---- outputs = go1 + go0[..., 16:] ----
__global__ void k_outsum(const float* go1, const float* go0, float* outF) {
  int idx = blockIdx.x * blockDim.x + threadIdx.x;
  if (idx >= BB * RCc * NN * 16) return;
  int t = idx & 15;
  int rest = idx >> 4;
  outF[idx] = go1[idx] + go0[(size_t)rest * TT + 16 + t];
}

__global__ void k_end1(const float* outF, const float* W, const float* bv, float* out1) {
  int idx = blockIdx.x * blockDim.x + threadIdx.x;
  if (idx >= BB * HH * NN * 16) return;
  int t = idx & 15;
  int n = (idx >> 4) % NN;
  int o = (idx / (16 * NN)) % HH;
  int b = idx / (16 * NN * HH);
  float s = bv[o];
  for (int c = 0; c < RCc; ++c)
    s += outF[((size_t)(b * RCc + c) * NN + n) * 16 + t] * W[o * RCc + c];
  out1[idx] = s;
}

__global__ void k_end2(const float* out1, const float* W, const float* bv, float* out2) {
  int idx = blockIdx.x * blockDim.x + threadIdx.x;
  if (idx >= BB * TT * NN * 16) return;
  int t = idx & 15;
  int n = (idx >> 4) % NN;
  int o = (idx / (16 * NN)) % TT;
  int b = idx / (16 * NN * TT);
  float s = bv[o];
  for (int c = 0; c < HH; ++c)
    s += out1[((size_t)(b * HH + c) * NN + n) * 16 + t] * W[o * HH + c];
  out2[idx] = s;
}

__global__ void k_end3(const float* out2, const float* Wlo, const float* blo, float* out3) {
  int idx = blockIdx.x * blockDim.x + threadIdx.x;
  if (idx >= BB * TT * NN) return;
  int n = idx % NN;
  int c = (idx / NN) % TT;
  int b = idx / (NN * TT);
  float s = blo[0];
  for (int t = 0; t < 16; ++t)
    s += out2[((size_t)(b * TT + c) * NN + n) * 16 + t] * Wlo[t];
  out3[idx] = s;
}

__global__ void k_f1(const float* out3, const float* y, const float* ym,
                     const float* W, const float* bv, float* f1b) {
  int idx = blockIdx.x * blockDim.x + threadIdx.x;
  if (idx >= BB * TT * 128) return;
  int u = idx & 127;
  int c = (idx >> 7) % TT;
  int b = idx / (128 * TT);
  const float* w = W + u * (3 * NN);
  size_t row = (size_t)(b * TT + c) * NN;
  float s = bv[u];
  for (int j = 0; j < NN; ++j) s += out3[row + j] * w[j];
  for (int j = 0; j < NN; ++j) s += y[row + j] * w[NN + j];
  for (int j = 0; j < NN; ++j) s += ym[row + j] * w[2 * NN + j];
  f1b[idx] = s;
}

__global__ void k_f2(const float* f1b, const float* W, const float* bv, float* dout) {
  int idx = blockIdx.x * blockDim.x + threadIdx.x;
  if (idx >= BB * TT * NN) return;
  int v = idx % NN;
  int c = (idx / NN) % TT;
  int b = idx / (NN * TT);
  float s = bv[v];
  const float* f = f1b + (size_t)(b * TT + c) * 128;
  for (int u = 0; u < 128; ++u) s += f[u] * W[v * 128 + u];
  dout[(size_t)(b * TT + c) * NN + v] = s;
}

extern "C" void kernel_launch(void* const* d_in, const int* in_sizes, int n_in,
                              void* d_out, int out_size, void* d_ws, size_t ws_size,
                              hipStream_t stream) {
  (void)in_sizes; (void)n_in; (void)out_size; (void)ws_size;
  const float* inp    = (const float*)d_in[0];
  const float* msk    = (const float*)d_in[1];
  const float* inpbw  = (const float*)d_in[2];
  const float* mskbw  = (const float*)d_in[3];
  const float* yin    = (const float*)d_in[4];
  const float* ymask  = (const float*)d_in[5];
  const float* wsv    = (const float*)d_in[6];
  const float* bsv    = (const float*)d_in[7];
  const float* wcv    = (const float*)d_in[8];
  const float* bcv    = (const float*)d_in[9];
  const float* nv1    = (const float*)d_in[10];
  const float* nv2    = (const float*)d_in[11];
  const float* Wih_fw = (const float*)d_in[12];
  const float* Whh_fw = (const float*)d_in[13];
  const float* bih_fw = (const float*)d_in[14];
  const float* bhh_fw = (const float*)d_in[15];
  const float* Wih_bw = (const float*)d_in[16];
  const float* Whh_bw = (const float*)d_in[17];
  const float* bih_bw = (const float*)d_in[18];
  const float* bhh_bw = (const float*)d_in[19];
  const float* Wg     = (const float*)d_in[20];
  const float* bgv    = (const float*)d_in[21];
  const float* gamma  = (const float*)d_in[22];
  const float* beta   = (const float*)d_in[23];
  const float* W_end1 = (const float*)d_in[24];
  const float* b_end1 = (const float*)d_in[25];
  const float* W_end2 = (const float*)d_in[26];
  const float* b_end2 = (const float*)d_in[27];
  const float* W_lo   = (const float*)d_in[28];
  const float* b_lo   = (const float*)d_in[29];
  const float* W_f1   = (const float*)d_in[30];
  const float* b_f1   = (const float*)d_in[31];
  const float* W_f2   = (const float*)d_in[32];
  const float* b_f2   = (const float*)d_in[33];
  float* dout = (float*)d_out;

  char* wb = (char*)d_ws;
  size_t off = 0;
  auto alloc = [&](size_t bytes) -> void* {
    void* p = wb + off;
    off = (off + bytes + 255) & ~(size_t)255;
    return p;
  };
  __bf16* Wt   = (__bf16*)alloc((size_t)G4c * HNc * 2);   // tiled bf16 weights (reused 6x)
  float*  adp  = (float*)alloc((size_t)NN * NN * 4);
  __bf16* XinA = (__bf16*)alloc((size_t)256 * HNc * 2);
  __bf16* XinB = (__bf16*)alloc((size_t)256 * HNc * 2);
  __bf16* XinC = (__bf16*)alloc((size_t)128 * HNc * 2);
  float*  XpA  = (float*)alloc((size_t)256 * G4c * 4);
  float*  XpB  = (float*)alloc((size_t)256 * G4c * 4);
  float*  XpC  = (float*)alloc((size_t)128 * G4c * 4);
  __bf16* h0   = (__bf16*)alloc((size_t)BB * HNc * 2);
  __bf16* h1   = (__bf16*)alloc((size_t)BB * HNc * 2);
  float*  cst  = (float*)alloc((size_t)BB * HNc * 4);
  float*  hsA  = (float*)alloc((size_t)BB * HNc * TT * 4);
  float*  hsB  = (float*)alloc((size_t)BB * HNc * TT * 4);
  float*  hsC  = (float*)alloc((size_t)BB * HNc * 16 * 4);
  float*  gin  = (float*)alloc((size_t)BB * HNc * TT * 4);
  float*  x1b  = (float*)alloc((size_t)BB * HNc * TT * 4);
  float*  x2b  = (float*)alloc((size_t)BB * HNc * TT * 4);
  float*  go0  = (float*)alloc((size_t)BB * RCc * NN * TT * 4);
  float*  go1  = (float*)alloc((size_t)BB * RCc * NN * 16 * 4);
  float*  stats= (float*)alloc((size_t)2 * RCc * 4);
  float*  outF = (float*)alloc((size_t)BB * RCc * NN * 16 * 4);
  float*  out1 = (float*)alloc((size_t)BB * HH * NN * 16 * 4);
  float*  out2 = (float*)alloc((size_t)BB * TT * NN * 16 * 4);
  float*  out3 = (float*)alloc((size_t)BB * TT * NN * 4);
  float*  f1b  = (float*)alloc((size_t)BB * TT * 128 * 4);

  const int TPB = 256;
  auto nb = [](int total) { return (total + 255) / 256; };
  const int WCB = nb(JT * KT * 32);

  // adjacency + feature packing
  k_adp<<<1, 160, 0, stream>>>(nv1, nv2, adp);
  k_featpack<<<nb(BB * RCc * NN * TT), TPB, 0, stream>>>(inp, msk, wsv, bsv, wcv, bcv, XinA);
  k_featpack<<<nb(BB * RCc * NN * TT), TPB, 0, stream>>>(inpbw, mskbw, wsv, bsv, wcv, bcv, XinB);

  // batched input projections (WMMA bf16, tiled weights, convert-once each)
  k_wconv<<<WCB, TPB, 0, stream>>>(Wih_fw, Wt);
  k_xproj<<<dim3(16, 144), 128, 0, stream>>>(XinA, Wt, bih_fw, bhh_fw, XpA);
  k_wconv<<<WCB, TPB, 0, stream>>>(Wih_bw, Wt);
  k_xproj<<<dim3(16, 144), 128, 0, stream>>>(XinB, Wt, bih_bw, bhh_bw, XpB);

  // ---- layer 0 forward LSTM (Whh bf16 tiled, L2-resident across steps) ----
  k_wconv<<<WCB, TPB, 0, stream>>>(Whh_fw, Wt);
  hipMemsetAsync(h0, 0, (size_t)BB * HNc * 2, stream);
  hipMemsetAsync(cst, 0, (size_t)BB * HNc * 4, stream);
  for (int t = 0; t < TT; ++t)
    k_lstm_step<<<HNc / 16, 128, 0, stream>>>(Wt, XpA, cst,
        (t & 1) ? h1 : h0, (t & 1) ? h0 : h1, hsA, t, TT);

  // ---- layer 0 backward LSTM ----
  k_wconv<<<WCB, TPB, 0, stream>>>(Whh_bw, Wt);
  hipMemsetAsync(h0, 0, (size_t)BB * HNc * 2, stream);
  hipMemsetAsync(cst, 0, (size_t)BB * HNc * 4, stream);
  for (int t = 0; t < TT; ++t)
    k_lstm_step<<<HNc / 16, 128, 0, stream>>>(Wt, XpB, cst,
        (t & 1) ? h1 : h0, (t & 1) ? h0 : h1, hsB, t, TT);

  // combine + tanh, graph conv layer 0
  k_gcnin0<<<nb(BB * HNc * TT), TPB, 0, stream>>>(hsA, hsB, gin);
  k_diffuse<<<nb(BB * HH * NN * TT), TPB, 0, stream>>>(gin, adp, x1b, TT);
  k_diffuse<<<nb(BB * HH * NN * TT), TPB, 0, stream>>>(x1b, adp, x2b, TT);
  k_gcmix<<<nb(BB * RCc * NN * TT), TPB, 0, stream>>>(gin, x1b, x2b, Wg, bgv, go0, TT);

  // bn(go0[..., ::2]) -> packed layer-1 input
  k_bnstats<<<RCc, 256, 0, stream>>>(go0, stats);
  k_bnpack<<<nb(BB * RCc * NN * 16), TPB, 0, stream>>>(go0, stats, gamma, beta, XinC);

  // layer-1 projection + LSTM
  k_wconv<<<WCB, TPB, 0, stream>>>(Wih_fw + (size_t)G4c * HNc, Wt);
  k_xproj<<<dim3(8, 144), 128, 0, stream>>>(XinC, Wt, bih_fw + G4c, bhh_fw + G4c, XpC);
  k_wconv<<<WCB, TPB, 0, stream>>>(Whh_fw + (size_t)G4c * HNc, Wt);
  hipMemsetAsync(h0, 0, (size_t)BB * HNc * 2, stream);
  hipMemsetAsync(cst, 0, (size_t)BB * HNc * 4, stream);
  for (int t = 0; t < 16; ++t)
    k_lstm_step<<<HNc / 16, 128, 0, stream>>>(Wt, XpC, cst,
        (t & 1) ? h1 : h0, (t & 1) ? h0 : h1, hsC, t, 16);

  // graph conv layer 1
  k_gcnin1<<<nb(BB * HNc * 16), TPB, 0, stream>>>(hsC, gin);
  k_diffuse<<<nb(BB * HH * NN * 16), TPB, 0, stream>>>(gin, adp, x1b, 16);
  k_diffuse<<<nb(BB * HH * NN * 16), TPB, 0, stream>>>(x1b, adp, x2b, 16);
  k_gcmix<<<nb(BB * RCc * NN * 16), TPB, 0, stream>>>(gin, x1b, x2b,
      Wg + RCc * 48, bgv + RCc, go1, 16);

  // residual accumulate + head
  k_outsum<<<nb(BB * RCc * NN * 16), TPB, 0, stream>>>(go1, go0, outF);
  k_end1<<<nb(BB * HH * NN * 16), TPB, 0, stream>>>(outF, W_end1, b_end1, out1);
  k_end2<<<nb(BB * TT * NN * 16), TPB, 0, stream>>>(out1, W_end2, b_end2, out2);
  k_end3<<<nb(BB * TT * NN), TPB, 0, stream>>>(out2, W_lo, b_lo, out3);
  k_f1<<<nb(BB * TT * 128), TPB, 0, stream>>>(out3, yin, ymask, W_f1, b_f1, f1b);
  k_f2<<<nb(BB * TT * NN), TPB, 0, stream>>>(f1b, W_f2, b_f2, dout);
}
